// FluxObserverHiPPO_66657892434506
// MI455X (gfx1250) — matrix operand order repaired and checked by reading, not compiled, not run on hardware
//
#include <hip/hip_runtime.h>
#include <hip/hip_bf16.h>

typedef __attribute__((ext_vector_type(16))) _Float16 v16h;
typedef __attribute__((ext_vector_type(8)))  float    v8f;

#define NST     8
#define BATCH   32
#define TLEN    8192
#define NCH     2
#define CHUNK   32
#define NCHUNK  (TLEN / CHUNK)      // 256
#define NB      13                  // GRID + K spline bases per edge
#define NTOK    (BATCH * TLEN)      // 262144

// ---------------- workspace layout (bytes) ----------------
#define OFF_H  ((size_t)0)                                  // h: [NTOK][16] f32 = 16 MB
#define OFF_Q  (OFF_H + (size_t)NTOK * 16 * 4)              // Q: [NCHUNK][8][8] f32
#define OFF_U  (OFF_Q + (size_t)NCHUNK * 64 * 4)            // U: [NCHUNK][64][8] f32
#define OFF_S  (OFF_U + (size_t)NCHUNK * 64 * 8 * 4)        // S: [NCHUNK][64][8] f32
#define OFF_PK (OFF_S + (size_t)NCHUNK * 64 * 8 * 4)        // packed f16 weights
// packed-weight offsets in halves (tile = 32 lanes * 16 halves = 512)
#define PK_WB1 0        // 1 K-tile  x 2 N-tiles
#define PK_WS1 1024     // 7 K-tiles x 2 N-tiles
#define PK_WB2 8192     // 1 x 1
#define PK_WS2 8704     // 13 x 1

__constant__ float c_r[8] = {1.0f, 1.73205081f, 2.23606798f, 2.64575131f,
                             3.0f, 3.31662479f, 3.60555128f, 3.87298335f};

__device__ __forceinline__ float silu_f(float x) { return x / (1.0f + __expf(-x)); }

// Cox-de Boor on the shared uniform grid (GRID=10, K=3, h=0.2) -> 13 bases
__device__ __forceinline__ void bsplines13(float x, float bb[NB]) {
  const float H = 0.2f;
  float b[16];
#pragma unroll
  for (int i = 0; i < 16; i++) {
    float g0 = (float)(i - 3) * H - 1.0f;
    float g1 = (float)(i - 2) * H - 1.0f;
    b[i] = (x >= g0 && x < g1) ? 1.0f : 0.0f;
  }
#pragma unroll
  for (int p = 1; p <= 3; p++) {
    float inv = 1.0f / ((float)p * H);
#pragma unroll
    for (int i = 0; i < 16 - p; i++) {
      float gi   = (float)(i - 3) * H - 1.0f;
      float gip1 = (float)(i - 2 + p) * H - 1.0f;
      b[i] = ((x - gi) * b[i] + (gip1 - x) * b[i + 1]) * inv;
    }
  }
#pragma unroll
  for (int q = 0; q < NB; q++) bb[q] = b[q];
}

// One HiPPO step: s <- (I - A/t) s + x * r/t ; A = tril(r r^T,-1) + diag(i+1)
__device__ __forceinline__ void hippo_step(float s[NST], float x, float inv_t) {
  float acc = 0.0f;
#pragma unroll
  for (int i = 0; i < NST; i++) {
    float ri   = c_r[i];
    float snew = s[i] - ((float)(i + 1) * s[i] + ri * acc) * inv_t + x * ri * inv_t;
    acc += ri * s[i];
    s[i] = snew;
  }
}

// ================= weight packing (f32 -> f16, WMMA B layout) =================
// B 32x16 f16 layout: lane<16: halves 0-7->K0-7, 8-15->K16-23; lane>=16: K8-15 / K24-31; N = lane%16
__device__ __forceinline__ void pack_one(const float* W, int inF, int outF, int KT, int NT,
                                         _Float16* dst, int tid, int nthr) {
  int total = KT * NT * 512;
  for (int e = tid; e < total; e += nthr) {
    int tile = e >> 9;
    int rem  = e & 511;
    int lane = rem >> 4;
    int i    = rem & 15;
    int kt = tile / NT, nt = tile % NT;
    int k = kt * 32 + ((i >> 3) << 4) + ((lane >> 4) << 3) + (i & 7);
    int n = nt * 16 + (lane & 15);
    float v = (k < inF && n < outF) ? W[n * inF + k] : 0.0f;
    dst[e] = (_Float16)v;
  }
}

__global__ void pack_weights_kernel(const float* wb1, const float* ws1,
                                    const float* wb2, const float* ws2, _Float16* pk) {
  int tid = threadIdx.x, nthr = blockDim.x;
  pack_one(wb1, 16,  32, 1,  2, pk + PK_WB1, tid, nthr);
  pack_one(ws1, 208, 32, 7,  2, pk + PK_WS1, tid, nthr);
  pack_one(wb2, 32,  16, 1,  1, pk + PK_WB2, tid, nthr);
  pack_one(ws2, 416, 16, 13, 1, pk + PK_WS2, tid, nthr);
}

// ================= HiPPO chunked parallel scan =================
// Pass 1: per chunk compute Q (8x8 transition) and u (input injection) per (b,c)
__global__ void hippo_partial_kernel(const float* __restrict__ x_seq,
                                     float* __restrict__ Q, float* __restrict__ U) {
  int chunk = blockIdx.x;
  int tid   = threadIdx.x;           // 0..63 -> (b,c)
  int b = tid >> 1, c = tid & 1;
  float u[NST];
  float q[NST];
#pragma unroll
  for (int i = 0; i < NST; i++) { u[i] = 0.0f; q[i] = (i == tid) ? 1.0f : 0.0f; }

  for (int l = 0; l < CHUNK; l++) {
    int   t     = chunk * CHUNK + l + 1;
    float inv_t = 1.0f / (float)t;
    float x     = x_seq[((size_t)b * TLEN + (t - 1)) * NCH + c];
    hippo_step(u, x, inv_t);
    if (tid < NST) hippo_step(q, 0.0f, inv_t);   // column tid of chunk matrix
  }
#pragma unroll
  for (int i = 0; i < NST; i++) U[((size_t)chunk * 64 + tid) * NST + i] = u[i];
  if (tid < NST) {
#pragma unroll
    for (int i = 0; i < NST; i++) Q[(size_t)chunk * 64 + i * NST + tid] = q[i];
  }
}

// Pass 2: tiny sequential combine over 256 chunks: S_{j+1} = Q_j S_j + u_j
__global__ void hippo_combine_kernel(const float* __restrict__ Q,
                                     const float* __restrict__ U,
                                     float* __restrict__ S) {
  __shared__ float qs[64];
  int tid = threadIdx.x;             // 0..63
  float s[NST];
#pragma unroll
  for (int i = 0; i < NST; i++) s[i] = 0.0f;
  for (int j = 0; j < NCHUNK; j++) {
#pragma unroll
    for (int i = 0; i < NST; i++) S[((size_t)j * 64 + tid) * NST + i] = s[i];
    qs[tid] = Q[(size_t)j * 64 + tid];
    __syncthreads();
    float sn[NST];
#pragma unroll
    for (int i = 0; i < NST; i++) {
      float a = U[((size_t)j * 64 + tid) * NST + i];
#pragma unroll
      for (int k = 0; k < NST; k++) a += qs[i * NST + k] * s[k];
      sn[i] = a;
    }
    __syncthreads();
#pragma unroll
    for (int i = 0; i < NST; i++) s[i] = sn[i];
  }
}

// Pass 3: replay each chunk from its start state and emit all h states
__global__ void hippo_emit_kernel(const float* __restrict__ x_seq,
                                  const float* __restrict__ S,
                                  float* __restrict__ h) {
  int chunk = blockIdx.x;
  int tid   = threadIdx.x;
  int b = tid >> 1, c = tid & 1;
  float s[NST];
#pragma unroll
  for (int i = 0; i < NST; i++) s[i] = S[((size_t)chunk * 64 + tid) * NST + i];
  for (int l = 0; l < CHUNK; l++) {
    int   t     = chunk * CHUNK + l + 1;
    float inv_t = 1.0f / (float)t;
    float x     = x_seq[((size_t)b * TLEN + (t - 1)) * NCH + c];
    hippo_step(s, x, inv_t);
    float4* p = (float4*)(h + ((size_t)b * TLEN + (t - 1)) * 16 + c * 8);
    float4 lo, hi2;
    lo.x = s[0]; lo.y = s[1]; lo.z = s[2]; lo.w = s[3];
    hi2.x = s[4]; hi2.y = s[5]; hi2.z = s[6]; hi2.w = s[7];
    p[0] = lo; p[1] = hi2;
  }
}

// ================= fused KAN (WMMA) =================
// 16-bit A 16x32 layout: lane<16: halves 0-7->K0-7, 8-15->K16-23; lane>=16: K8-15 / K24-31
__device__ __forceinline__ int a_kmap(int i, int lane) {
  return ((i >> 3) << 4) + ((lane >> 4) << 3) + (i & 7);
}

__device__ __forceinline__ v16h build_base_a(const float* act, int m, int lane, int F, int S) {
  v16h a;
#pragma unroll
  for (int i = 0; i < 16; i++) {
    int K = a_kmap(i, lane);
    float v = 0.0f;
    if (K < F) v = silu_f(act[m * S + K]);
    a[i] = (_Float16)v;
  }
  return a;
}

__device__ __forceinline__ v16h build_spline_a(const float* sp, int m, int lane,
                                               int FNB, int S, int kt) {
  v16h a;
#pragma unroll
  for (int i = 0; i < 16; i++) {
    int K = kt * 32 + a_kmap(i, lane);
    float v = 0.0f;
    if (K < FNB) v = sp[m * S + K];
    a[i] = (_Float16)v;
  }
  return a;
}

__device__ __forceinline__ v16h load_btile(const _Float16* pkbase, int tile, int lane) {
  const v16h* p = (const v16h*)pkbase;
  return p[tile * 32 + lane];
}

#define WMMA_F16(A, B, C) \
  __builtin_amdgcn_wmma_f32_16x16x32_f16(false, (A), false, (B), (short)0, (C), false, false)

__global__ __launch_bounds__(128) void kan_fused_kernel(const float* __restrict__ h,
                                                        const _Float16* __restrict__ pk,
                                                        const float* __restrict__ wb3,
                                                        const float* __restrict__ ws3,
                                                        float* __restrict__ out) {
  extern __shared__ float smem[];
  int tid  = threadIdx.x;
  int w    = tid >> 5;              // wave 0..3
  int lane = tid & 31;
  int m    = lane & 15;             // token row within 16-token tile
  int hi   = lane >> 4;

  float* s_w3 = smem;                                   // 224 floats
  float* base = smem + 224;
  float* s_a1 = base + w * 256;                         // [16][16]  layer1 in
  float* s_a2 = base + 4 * 256 + w * 512;               // [16][32]  layer1 out
  float* s_a3 = base + 4 * 256 + 4 * 512 + w * 256;     // [16][16]  layer2 out
  float* s_sp = base + 4 * 256 + 4 * 512 + 4 * 256 + w * 6656;  // [16][416] spline bases

  // layer-3 weights -> LDS
  if (tid < 16) s_w3[tid] = wb3[tid];
  for (int j = tid; j < 208; j += 128) s_w3[16 + j] = ws3[j];

  int tokBase = blockIdx.x * 128 + w * 16;
  __builtin_prefetch(h + (size_t)(tokBase + 64) * 16, 0, 1);   // -> global_prefetch_b8

  // stage h tile (16 tokens x 16 feats) into LDS
  {
    const float4* g = (const float4*)(h + (size_t)(tokBase + m) * 16 + hi * 8);
    float4 v0 = g[0], v1 = g[1];
    float4* d = (float4*)(s_a1 + m * 16 + hi * 8);
    d[0] = v0; d[1] = v1;
  }
  __syncthreads();

  // ---- layer 1 spline bases: 256 scalars/wave, 8 per lane ----
  for (int j = 0; j < 8; j++) {
    int sIdx = lane * 8 + j;
    int tok = sIdx >> 4, f = sIdx & 15;
    float bb[NB];
    bsplines13(s_a1[tok * 16 + f], bb);
#pragma unroll
    for (int q = 0; q < NB; q++) s_sp[tok * 208 + f * NB + q] = bb[q];
  }
  __syncthreads();

  // ---- layer 1 GEMMs: [16tok x (16 base | 208 spline)] -> 32 ----
  v8f c0, c1;
#pragma unroll
  for (int g = 0; g < 8; g++) { c0[g] = 0.0f; c1[g] = 0.0f; }
  {
    v16h a  = build_base_a(s_a1, m, lane, 16, 16);
    v16h b0 = load_btile(pk + PK_WB1, 0, lane);
    v16h b1 = load_btile(pk + PK_WB1, 1, lane);
    c0 = WMMA_F16(a, b0, c0);
    c1 = WMMA_F16(a, b1, c1);
  }
#pragma unroll
  for (int kt = 0; kt < 7; kt++) {
    v16h a  = build_spline_a(s_sp, m, lane, 208, 208, kt);
    v16h b0 = load_btile(pk + PK_WS1, kt * 2 + 0, lane);
    v16h b1 = load_btile(pk + PK_WS1, kt * 2 + 1, lane);
    c0 = WMMA_F16(a, b0, c0);
    c1 = WMMA_F16(a, b1, c1);
  }
  // C/D layout: lane<16 -> M=g, lane>=16 -> M=8+g; N = lane%16 (+16 for c1 tile)
  {
    int rbase = hi << 3;
#pragma unroll
    for (int g = 0; g < 8; g++) {
      int row = g + rbase;
      s_a2[row * 32 + (lane & 15)]      = c0[g];
      s_a2[row * 32 + (lane & 15) + 16] = c1[g];
    }
  }
  __syncthreads();

  // ---- layer 2 spline bases: 512 scalars/wave, 16 per lane ----
  for (int j = 0; j < 16; j++) {
    int sIdx = lane * 16 + j;
    int tok = sIdx >> 5, f = sIdx & 31;
    float bb[NB];
    bsplines13(s_a2[tok * 32 + f], bb);
#pragma unroll
    for (int q = 0; q < NB; q++) s_sp[tok * 416 + f * NB + q] = bb[q];
  }
  __syncthreads();

  // ---- layer 2 GEMMs: [16tok x (32 base | 416 spline)] -> 16 ----
  v8f d2;
#pragma unroll
  for (int g = 0; g < 8; g++) d2[g] = 0.0f;
  {
    v16h a = build_base_a(s_a2, m, lane, 32, 32);
    v16h b = load_btile(pk + PK_WB2, 0, lane);
    d2 = WMMA_F16(a, b, d2);
  }
#pragma unroll
  for (int kt = 0; kt < 13; kt++) {
    v16h a = build_spline_a(s_sp, m, lane, 416, 416, kt);
    v16h b = load_btile(pk + PK_WS2, kt, lane);
    d2 = WMMA_F16(a, b, d2);
  }
  {
    int rbase = hi << 3;
#pragma unroll
    for (int g = 0; g < 8; g++) s_a3[(g + rbase) * 16 + (lane & 15)] = d2[g];
  }
  __syncthreads();

  // ---- layer 3: 16 -> 1, VALU dot per token on lanes 0..15 ----
  if (lane < 16) {
    float r = 0.0f;
#pragma unroll
    for (int j = 0; j < 16; j++) {
      float y = s_a3[m * 16 + j];
      r += silu_f(y) * s_w3[j];
      float bb[NB];
      bsplines13(y, bb);
#pragma unroll
      for (int q = 0; q < NB; q++) r += bb[q] * s_w3[16 + j * NB + q];
    }
    out[tokBase + m] = r;
  }
}

// ================= launch =================
extern "C" void kernel_launch(void* const* d_in, const int* in_sizes, int n_in,
                              void* d_out, int out_size, void* d_ws, size_t ws_size,
                              hipStream_t stream) {
  (void)in_sizes; (void)n_in; (void)out_size; (void)ws_size;
  const float* x_seq = (const float*)d_in[0];
  const float* wb1   = (const float*)d_in[1];
  const float* ws1   = (const float*)d_in[2];
  const float* wb2   = (const float*)d_in[3];
  const float* ws2   = (const float*)d_in[4];
  const float* wb3   = (const float*)d_in[5];
  const float* ws3   = (const float*)d_in[6];

  char* ws = (char*)d_ws;
  float*     h  = (float*)(ws + OFF_H);
  float*     Q  = (float*)(ws + OFF_Q);
  float*     U  = (float*)(ws + OFF_U);
  float*     S  = (float*)(ws + OFF_S);
  _Float16*  pk = (_Float16*)(ws + OFF_PK);

  pack_weights_kernel<<<1, 256, 0, stream>>>(wb1, ws1, wb2, ws2, pk);
  hippo_partial_kernel<<<NCHUNK, 64, 0, stream>>>(x_seq, Q, U);
  hippo_combine_kernel<<<1, 64, 0, stream>>>(Q, U, S);
  hippo_emit_kernel<<<NCHUNK, 64, 0, stream>>>(x_seq, S, h);

  size_t smem_bytes = (size_t)(224 + 4 * (256 + 512 + 256 + 6656)) * sizeof(float);
  kan_fused_kernel<<<NTOK / 128, 128, smem_bytes, stream>>>(h, pk, wb3, ws3, (float*)d_out);
}
